// SelectiveDLinOSSLayer_31499290148991
// MI455X (gfx1250) — matrix operand order repaired and checked by reading, not compiled, not run on hardware
//
#include <hip/hip_runtime.h>
#include <hip/hip_bf16.h>

#define BATCH 16
#define LSEQ  4096
#define HDIM  256
#define MDIM  256
#define BLROWS (BATCH*LSEQ)      // 65536
#define CH    64                 // scan chunk length
#define NCH   (LSEQ/CH)          // 64 chunks
#define PIF 3.14159265358979f

typedef __attribute__((ext_vector_type(16))) __bf16 v16bf;
typedef __attribute__((ext_vector_type(8)))  __bf16 v8bf;
typedef __attribute__((ext_vector_type(4)))  __bf16 v4bf;
typedef __attribute__((ext_vector_type(8)))  float  v8f;

__device__ __forceinline__ float fast_sigmoid(float x) { return 1.0f / (1.0f + __expf(-x)); }
__device__ __forceinline__ float silu_f(float x)       { return x * fast_sigmoid(x); }

// ---- Tensor Data Mover tile loader (16-bit elements, 2D tile) ---------------
#if __has_builtin(__builtin_amdgcn_tensor_load_to_lds) && __has_builtin(__builtin_amdgcn_s_wait_tensorcnt)
#define HAVE_TDM 1
typedef __attribute__((ext_vector_type(4))) unsigned int u32x4;
typedef __attribute__((ext_vector_type(8))) int          i32x8;
typedef __attribute__((ext_vector_type(4))) int          i32x4;

// D# per CDNA5 ISA: group0 = {flags, lds_addr, global_addr, type=2},
// group1 = {data_size=2B, tensor_dim0/1, tile_dim0/1, tensor_dim0_stride}.
__device__ __forceinline__ void tdm_load_2d_bf16(const __bf16* gsrc, __bf16* lds_dst,
                                                 unsigned rows, unsigned cols,
                                                 unsigned row_stride_elems) {
  unsigned long long ga = (unsigned long long)(uintptr_t)gsrc;
  unsigned int lds_off  = (unsigned int)(uintptr_t)lds_dst;   // LDS offset = addr[31:0]
  u32x4 g0;
  g0[0] = 1u;                                                  // count=1, user descriptor
  g0[1] = lds_off;                                             // lds_addr (bits 63:32)
  g0[2] = (unsigned int)(ga & 0xFFFFFFFFull);                  // global_addr[31:0]
  g0[3] = (unsigned int)((ga >> 32) & 0x1FFFFFFull) | (2u << 30); // global_addr[56:32] | type=2
  i32x8 g1;
  g1[0] = (int)(1u << 16);                                     // workgroup_mask=0, data_size=1 (2B)
  g1[1] = (int)((cols & 0xFFFFu) << 16);                       // tensor_dim0[15:0]
  g1[2] = (int)((cols >> 16) & 0xFFFFu);                       // tensor_dim0[31:16] | tensor_dim1[15:0]=0
  g1[3] = (int)(16u | (cols << 16));                           // tensor_dim1[31:16]=16 (1M rows) | tile_dim0
  g1[4] = (int)rows;                                           // tile_dim1=rows, tile_dim2=0
  g1[5] = (int)row_stride_elems;                               // tensor_dim0_stride[31:0]
  g1[6] = 0; g1[7] = 0;
  i32x4 z4 = {0, 0, 0, 0};
  i32x8 z8 = {0, 0, 0, 0, 0, 0, 0, 0};
  __builtin_amdgcn_tensor_load_to_lds(g0, g1, z4, z4, z8, 0);  // 6-arg (clang-23) form
}
#else
#define HAVE_TDM 0
#endif

// ---- WMMA fragment helpers (bf16, 16x16x32, wave32) -------------------------
__device__ __forceinline__ v16bf load_a_frag(const __bf16* base, int stride, int k0, int lane) {
  int r    = lane & 15;
  int koff = (lane >> 4) << 3;
  const __bf16* p = base + r * stride + k0 + koff;
  v8bf lo = *(const v8bf*)(p);
  v8bf hi = *(const v8bf*)(p + 16);
  v16bf out;
#pragma unroll
  for (int i = 0; i < 8; ++i) { out[i] = lo[i]; out[i + 8] = hi[i]; }
  return out;
}
__device__ __forceinline__ v16bf load_b_frag(const __bf16* w, int stride, int n0, int k0, int lane) {
  int n  = n0 + (lane & 15);
  int kb = k0 + ((lane >> 4) << 3);
  const __bf16* p = w + n * stride + kb;
  v8bf lo = *(const v8bf*)(p);
  v8bf hi = *(const v8bf*)(p + 16);
  v16bf out;
#pragma unroll
  for (int i = 0; i < 8; ++i) { out[i] = lo[i]; out[i + 8] = hi[i]; }
  return out;
}
__device__ __forceinline__ v8f wmma_bf16(v16bf a, v16bf b, v8f c) {
  return __builtin_amdgcn_wmma_f32_16x16x32_bf16(false, a, false, b, (short)0, c, false, false);
}

// ---- Kernel 1a: inputs f32 -> bf16 -----------------------------------------
__global__ __launch_bounds__(256) void cvt_inputs_kernel(const float* __restrict__ in,
                                                         __bf16* __restrict__ out) {
  int i = (blockIdx.x * 256 + threadIdx.x) * 4;
  float4 v = *(const float4*)(in + i);
  v4bf o = { (__bf16)v.x, (__bf16)v.y, (__bf16)v.z, (__bf16)v.w };
  *(v4bf*)(out + i) = o;
}

// ---- Kernel 1b: weights -> bf16 (C_im pre-negated) -------------------------
__global__ __launch_bounds__(256) void cvt_weights_kernel(
    const float* __restrict__ encw, const float* __restrict__ rw,
    const float* __restrict__ thw,  const float* __restrict__ injw,
    const float* __restrict__ Bp,   const float* __restrict__ Cp,
    __bf16* __restrict__ wbf) {
  int i = blockIdx.x * 256 + threadIdx.x;        // 0..65535
  wbf[0 * 65536 + i] = (__bf16)encw[i];
  wbf[1 * 65536 + i] = (__bf16)rw[i];
  wbf[2 * 65536 + i] = (__bf16)thw[i];
  wbf[3 * 65536 + i] = (__bf16)injw[i];
  wbf[4 * 65536 + i] = (__bf16)Bp[2 * i + 0];    // B_re (M,H)
  wbf[5 * 65536 + i] = (__bf16)Bp[2 * i + 1];    // B_im (M,H)
  wbf[6 * 65536 + i] = (__bf16)Cp[2 * i + 0];    // C_re (H,M)
  wbf[7 * 65536 + i] = (__bf16)(-Cp[2 * i + 1]); // -C_im (H,M)
}

// ---- Kernel 2: encoder GEMM + bias + SiLU ----------------------------------
__global__ __launch_bounds__(256) void enc_gemm_kernel(const __bf16* __restrict__ inbf,
                                                       const __bf16* __restrict__ encw,
                                                       const float* __restrict__ enc_b,
                                                       float* __restrict__ enc_out) {
  __shared__ __align__(16) __bf16 atile[16 * HDIM];            // 8 KB
  int row0 = blockIdx.x * 16;
#if HAVE_TDM
  if ((threadIdx.x >> 5) == 0)
    tdm_load_2d_bf16(inbf + row0 * HDIM, atile, 16, HDIM, HDIM);
  __builtin_amdgcn_s_wait_tensorcnt(0);
#else
  {
    const uint4* s = (const uint4*)(inbf + row0 * HDIM);
    uint4* d = (uint4*)atile;
    for (int c = threadIdx.x; c < 16 * HDIM / 8; c += 256) d[c] = s[c];
  }
#endif
  __syncthreads();
  int wave = threadIdx.x >> 5, lane = threadIdx.x & 31;
  int n0a = (2 * wave) * 16, n0b = (2 * wave + 1) * 16;
  v8f acc0 = {}, acc1 = {};
#pragma unroll
  for (int k0 = 0; k0 < HDIM; k0 += 32) {
    __builtin_prefetch(encw + n0a * HDIM + k0 + 64, 0, 0);
    v16bf a  = load_a_frag(atile, HDIM, k0, lane);
    v16bf b0 = load_b_frag(encw, HDIM, n0a, k0, lane);
    v16bf b1 = load_b_frag(encw, HDIM, n0b, k0, lane);
    acc0 = wmma_bf16(a, b0, acc0);
    acc1 = wmma_bf16(a, b1, acc1);
  }
  int col = lane & 15, rsel = (lane >> 4) * 8;
  float bias0 = enc_b[n0a + col], bias1 = enc_b[n0b + col];
#pragma unroll
  for (int v = 0; v < 8; ++v) {
    int r = row0 + rsel + v;
    enc_out[r * HDIM + n0a + col] = silu_f(acc0[v] + bias0);
    enc_out[r * HDIM + n0b + col] = silu_f(acc1[v] + bias1);
  }
}

// ---- Kernel 3: depthwise causal conv (K=4) + SiLU -> bf16 feats ------------
__global__ __launch_bounds__(256) void conv_silu_kernel(const float* __restrict__ enc_out,
                                                        const float* __restrict__ conv_w,
                                                        const float* __restrict__ conv_b,
                                                        __bf16* __restrict__ feats) {
  int idx = blockIdx.x * 256 + threadIdx.x;      // over BL*H
  int h = idx & (HDIM - 1);
  int bl = idx >> 8;
  int l = bl & (LSEQ - 1);
  float w0 = conv_w[h * 4 + 0], w1 = conv_w[h * 4 + 1];
  float w2 = conv_w[h * 4 + 2], w3 = conv_w[h * 4 + 3];
  const float* p = enc_out + idx;
  float acc = conv_b[h] + w3 * p[0];
  if (l >= 1) acc += w2 * p[-HDIM];
  if (l >= 2) acc += w1 * p[-2 * HDIM];
  if (l >= 3) acc += w0 * p[-3 * HDIM];
  feats[idx] = (__bf16)silu_f(acc);
}

// ---- Kernel 4: fused 5-head GEMM + spectral conditioning -------------------
__global__ __launch_bounds__(256) void heads_gemm_kernel(
    const __bf16* __restrict__ feats, const __bf16* __restrict__ inbf,
    const __bf16* __restrict__ wbf,
    const float* __restrict__ r_base, const float* __restrict__ th_base,
    const float* __restrict__ dt_base,
    float* __restrict__ Aarr, float* __restrict__ Garr,
    float* __restrict__ bur, float* __restrict__ bui) {
  __shared__ __align__(16) __bf16 ftile[16 * HDIM];            // 8 KB
  __shared__ __align__(16) __bf16 itile[16 * HDIM];            // 8 KB
  int row0 = blockIdx.x * 16;
#if HAVE_TDM
  if ((threadIdx.x >> 5) == 0) {
    tdm_load_2d_bf16(feats + row0 * HDIM, ftile, 16, HDIM, HDIM);
    tdm_load_2d_bf16(inbf  + row0 * HDIM, itile, 16, HDIM, HDIM);
  }
  __builtin_amdgcn_s_wait_tensorcnt(0);
#else
  {
    const uint4* fs = (const uint4*)(feats + row0 * HDIM);
    const uint4* is = (const uint4*)(inbf + row0 * HDIM);
    uint4* fd = (uint4*)ftile; uint4* id = (uint4*)itile;
    for (int c = threadIdx.x; c < 16 * HDIM / 8; c += 256) { fd[c] = fs[c]; id[c] = is[c]; }
  }
#endif
  __syncthreads();
  int wave = threadIdx.x >> 5, lane = threadIdx.x & 31;
  int m0 = (blockIdx.y * 8 + wave) * 16;
  const __bf16* rw  = wbf + 1 * 65536;
  const __bf16* thw = wbf + 2 * 65536;
  const __bf16* gw  = wbf + 3 * 65536;
  const __bf16* b0w = wbf + 4 * 65536;
  const __bf16* b1w = wbf + 5 * 65536;
  v8f accr = {}, accth = {}, accg = {}, accbr = {}, accbi = {};
#pragma unroll
  for (int k0 = 0; k0 < HDIM; k0 += 32) {
    __builtin_prefetch(rw + m0 * HDIM + k0 + 64, 0, 0);
    v16bf fa = load_a_frag(ftile, HDIM, k0, lane);
    v16bf ia = load_a_frag(itile, HDIM, k0, lane);
    accr  = wmma_bf16(fa, load_b_frag(rw,  HDIM, m0, k0, lane), accr);
    accth = wmma_bf16(fa, load_b_frag(thw, HDIM, m0, k0, lane), accth);
    accg  = wmma_bf16(fa, load_b_frag(gw,  HDIM, m0, k0, lane), accg);
    accbr = wmma_bf16(ia, load_b_frag(b0w, HDIM, m0, k0, lane), accbr);
    accbi = wmma_bf16(ia, load_b_frag(b1w, HDIM, m0, k0, lane), accbi);
  }
  int col = lane & 15, rsel = (lane >> 4) * 8;
  int m = m0 + col;
  float rb  = r_base[m], tb = th_base[m];
  float dtc = fmaxf(fast_sigmoid(dt_base[m]), 1e-6f);
  int b = row0 >> 12;
  int l0 = row0 & (LSEQ - 1);
#pragma unroll
  for (int v = 0; v < 8; ++v) {
    int l = l0 + rsel + v;
    float r   = fast_sigmoid(rb + accr[v]);
    float th  = PIF * tanhf(tb + accth[v]);
    float r2  = fmaxf(r * r, 1e-8f);
    float Av  = fmaxf((r2 - 2.0f * r * __cosf(th) + 1.0f) / (dtc * dtc * r2), 0.0f);
    float Gv  = fmaxf((1.0f - r2) / (dtc * r2), 0.0f);
    float gate = fast_sigmoid(accg[v]);
    int idx = (l * BATCH + b) * MDIM + m;        // time-major for the scan
    Aarr[idx] = Av;
    Garr[idx] = Gv;
    bur[idx]  = accbr[v] * gate;
    bui[idx]  = accbi[v] * gate;
  }
}

// ---- Kernels 5a/5b/5c: chunked parallel scan over L ------------------------
// step: s' = M_t s + c_t with M_t = [[1/S, -q],[p, 1-dtc*q]], q = dtc*A/S, p = dtc/S
__global__ __launch_bounds__(256) void scan_pass1_kernel(
    const float* __restrict__ Aarr, const float* __restrict__ Garr,
    const float* __restrict__ bur,  const float* __restrict__ bui,
    const float* __restrict__ dt_base, float4* __restrict__ Msum, float4* __restrict__ Csum) {
  int b = blockIdx.x, ch = blockIdx.y, m = threadIdx.x;
  float dtc = fmaxf(fast_sigmoid(dt_base[m]), 1e-6f);
  float ma = 1.f, mb = 0.f, mc = 0.f, md = 1.f;
  float cr0 = 0.f, cr1 = 0.f, ci0 = 0.f, ci1 = 0.f;
  int base = b * MDIM + m;
  for (int i = 0; i < CH; ++i) {
    int idx = (ch * CH + i) * (BATCH * MDIM) + base;
    float Av = Aarr[idx], Gv = Garr[idx], br = bur[idx], bi = bui[idx];
    float iS = 1.f / fmaxf(1.f + dtc * Gv, 1e-6f);
    float p = dtc * iS, q = dtc * Av * iS, e = 1.f - dtc * q;
    float na = iS * ma - q * mc, nb = iS * mb - q * md;
    float nc = p * ma + e * mc,  nd = p * mb + e * md;
    ma = na; mb = nb; mc = nc; md = nd;
    float t0 = iS * cr0 - q * cr1 + p * br;
    float t1 = p * cr0 + e * cr1 + dtc * p * br;
    cr0 = t0; cr1 = t1;
    t0 = iS * ci0 - q * ci1 + p * bi;
    t1 = p * ci0 + e * ci1 + dtc * p * bi;
    ci0 = t0; ci1 = t1;
  }
  int o = (ch * BATCH + b) * MDIM + m;
  Msum[o] = make_float4(ma, mb, mc, md);
  Csum[o] = make_float4(cr0, cr1, ci0, ci1);
}

__global__ __launch_bounds__(256) void scan_pass2_kernel(
    const float4* __restrict__ Msum, const float4* __restrict__ Csum,
    float4* __restrict__ S0) {
  int b = blockIdx.x, m = threadIdx.x;
  float zr = 0.f, xr = 0.f, zi = 0.f, xi = 0.f;
  for (int ch = 0; ch < NCH; ++ch) {
    int o = (ch * BATCH + b) * MDIM + m;
    S0[o] = make_float4(zr, xr, zi, xi);
    float4 Mv = Msum[o], Cv = Csum[o];
    float nzr = Mv.x * zr + Mv.y * xr + Cv.x;
    float nxr = Mv.z * zr + Mv.w * xr + Cv.y;
    float nzi = Mv.x * zi + Mv.y * xi + Cv.z;
    float nxi = Mv.z * zi + Mv.w * xi + Cv.w;
    zr = nzr; xr = nxr; zi = nzi; xi = nxi;
  }
}

__global__ __launch_bounds__(256) void scan_pass3_kernel(
    const float* __restrict__ Aarr, const float* __restrict__ Garr,
    const float* __restrict__ bur,  const float* __restrict__ bui,
    const float* __restrict__ dt_base, const float4* __restrict__ S0,
    __bf16* __restrict__ xsre, __bf16* __restrict__ xsim) {
  int b = blockIdx.x, ch = blockIdx.y, m = threadIdx.x;
  float dtc = fmaxf(fast_sigmoid(dt_base[m]), 1e-6f);
  float4 s = S0[(ch * BATCH + b) * MDIM + m];
  float zr = s.x, xr = s.y, zi = s.z, xi = s.w;
  int base = b * MDIM + m;
  for (int i = 0; i < CH; ++i) {
    int idx = (ch * CH + i) * (BATCH * MDIM) + base;
    float Av = Aarr[idx], Gv = Garr[idx], br = bur[idx], bi = bui[idx];
    float inv = 1.f / fmaxf(1.f + dtc * Gv, 1e-6f);
    zr = (zr + dtc * (-Av * xr + br)) * inv;
    zi = (zi + dtc * (-Av * xi + bi)) * inv;
    xr += dtc * zr;
    xi += dtc * zi;
    xsre[idx] = (__bf16)xr;
    xsim[idx] = (__bf16)xi;
  }
}

// ---- Kernel 6: output GEMM (xs_re*C0 + xs_im*(-C1)) + D*inputs -------------
__global__ __launch_bounds__(256) void out_gemm_kernel(
    const __bf16* __restrict__ xsre, const __bf16* __restrict__ xsim,
    const __bf16* __restrict__ wbf,  const float* __restrict__ Dv,
    const float* __restrict__ inputs, float* __restrict__ out) {
  __shared__ __align__(16) __bf16 rtile[16 * MDIM];            // 8 KB
  __shared__ __align__(16) __bf16 itile[16 * MDIM];            // 8 KB
  int bb = blockIdx.x >> 8;                      // batch
  int l0 = (blockIdx.x & 255) * 16;              // L tile
#if HAVE_TDM
  if ((threadIdx.x >> 5) == 0) {
    // strided rows: row stride = BATCH*MDIM elements in time-major xs
    tdm_load_2d_bf16(xsre + (l0 * BATCH + bb) * MDIM, rtile, 16, MDIM, BATCH * MDIM);
    tdm_load_2d_bf16(xsim + (l0 * BATCH + bb) * MDIM, itile, 16, MDIM, BATCH * MDIM);
  }
  __builtin_amdgcn_s_wait_tensorcnt(0);
#else
  for (int c = threadIdx.x; c < 16 * MDIM / 8; c += 256) {
    int r = c >> 5, cc = c & 31;
    int src = ((l0 + r) * BATCH + bb) * MDIM;
    ((uint4*)rtile)[c] = *((const uint4*)(xsre + src) + cc);
    ((uint4*)itile)[c] = *((const uint4*)(xsim + src) + cc);
  }
#endif
  __syncthreads();
  int wave = threadIdx.x >> 5, lane = threadIdx.x & 31;
  const __bf16* c0w = wbf + 6 * 65536;
  const __bf16* c1w = wbf + 7 * 65536;           // pre-negated C_im
  int h0a = (2 * wave) * 16, h0b = (2 * wave + 1) * 16;
  v8f acc0 = {}, acc1 = {};
#pragma unroll
  for (int k0 = 0; k0 < MDIM; k0 += 32) {
    __builtin_prefetch(c0w + h0a * MDIM + k0 + 64, 0, 0);
    v16bf ra = load_a_frag(rtile, MDIM, k0, lane);
    v16bf ia = load_a_frag(itile, MDIM, k0, lane);
    acc0 = wmma_bf16(ra, load_b_frag(c0w, MDIM, h0a, k0, lane), acc0);
    acc0 = wmma_bf16(ia, load_b_frag(c1w, MDIM, h0a, k0, lane), acc0);
    acc1 = wmma_bf16(ra, load_b_frag(c0w, MDIM, h0b, k0, lane), acc1);
    acc1 = wmma_bf16(ia, load_b_frag(c1w, MDIM, h0b, k0, lane), acc1);
  }
  int col = lane & 15, rsel = (lane >> 4) * 8;
  float d0 = Dv[h0a + col], d1 = Dv[h0b + col];
#pragma unroll
  for (int v = 0; v < 8; ++v) {
    int l = l0 + rsel + v;
    int idx0 = (bb * LSEQ + l) * HDIM;
    out[idx0 + h0a + col] = acc0[v] + d0 * inputs[idx0 + h0a + col];
    out[idx0 + h0b + col] = acc1[v] + d1 * inputs[idx0 + h0b + col];
  }
}

// ---- Launch ----------------------------------------------------------------
extern "C" void kernel_launch(void* const* d_in, const int* in_sizes, int n_in,
                              void* d_out, int out_size, void* d_ws, size_t ws_size,
                              hipStream_t stream) {
  (void)in_sizes; (void)n_in; (void)out_size; (void)ws_size;
  const float* inputs   = (const float*)d_in[0];
  const float* B_param  = (const float*)d_in[1];
  const float* C_param  = (const float*)d_in[2];
  const float* Dv       = (const float*)d_in[3];
  const float* enc_w    = (const float*)d_in[4];
  const float* enc_b    = (const float*)d_in[5];
  const float* conv_w   = (const float*)d_in[6];
  const float* conv_b   = (const float*)d_in[7];
  const float* r_base   = (const float*)d_in[8];
  const float* th_base  = (const float*)d_in[9];
  const float* r_head_w = (const float*)d_in[10];
  const float* th_head_w= (const float*)d_in[11];
  const float* dt_base  = (const float*)d_in[12];
  const float* inj_w    = (const float*)d_in[13];
  float* out = (float*)d_out;

  char* ws = (char*)d_ws;
  const size_t MB = 1ull << 20;
  __bf16* wbf    = (__bf16*)(ws);                 //   1 MB : 8 bf16 weight mats
  __bf16* inbf   = (__bf16*)(ws + 1 * MB);        //  32 MB : inputs bf16
  __bf16* feats  = (__bf16*)(ws + 33 * MB);       //  32 MB : conv output bf16
  float*  encout = (float*)(ws + 65 * MB);        //  64 MB : encoder out f32
  float*  Garr   = (float*)(ws + 129 * MB);       //  64 MB
  float*  burv   = (float*)(ws + 193 * MB);       //  64 MB
  float*  buiv   = (float*)(ws + 257 * MB);       //  64 MB
  float4* Msum   = (float4*)(ws + 321 * MB);      //   4 MB : chunk transfer matrices
  float4* Csum   = (float4*)(ws + 325 * MB);      //   4 MB : chunk affine offsets
  float4* S0v    = (float4*)(ws + 329 * MB);      //   4 MB : chunk initial states
  float*  Aarr   = encout;                        // reuse (encout consumed by conv)
  __bf16* xsre   = inbf;                          // reuse (inbf consumed by heads)
  __bf16* xsim   = feats;                         // reuse (feats consumed by heads)

  // 1. precision conversion
  cvt_inputs_kernel<<<BLROWS * HDIM / 4 / 256, 256, 0, stream>>>(inputs, inbf);
  cvt_weights_kernel<<<65536 / 256, 256, 0, stream>>>(enc_w, r_head_w, th_head_w,
                                                      inj_w, B_param, C_param, wbf);
  // 2. encoder GEMM + SiLU
  enc_gemm_kernel<<<BLROWS / 16, 256, 0, stream>>>(inbf, wbf, enc_b, encout);
  // 3. depthwise causal conv + SiLU
  conv_silu_kernel<<<BLROWS * HDIM / 256, 256, 0, stream>>>(encout, conv_w, conv_b, feats);
  // 4. fused 5-head GEMM + spectral conditioning (time-major outputs)
  heads_gemm_kernel<<<dim3(BLROWS / 16, 2), 256, 0, stream>>>(
      feats, inbf, wbf, r_base, th_base, dt_base, Aarr, Garr, burv, buiv);
  // 5. chunked parallel scan (compose -> summary scan -> replay)
  scan_pass1_kernel<<<dim3(BATCH, NCH), MDIM, 0, stream>>>(
      Aarr, Garr, burv, buiv, dt_base, Msum, Csum);
  scan_pass2_kernel<<<BATCH, MDIM, 0, stream>>>(Msum, Csum, S0v);
  scan_pass3_kernel<<<dim3(BATCH, NCH), MDIM, 0, stream>>>(
      Aarr, Garr, burv, buiv, dt_base, S0v, xsre, xsim);
  // 6. output projection + skip
  out_gemm_kernel<<<BATCH * (LSEQ / 16), 256, 0, stream>>>(xsre, xsim, wbf, Dv, inputs, out);
}
